// EncoderDecoderRNN_74079595921939
// MI455X (gfx1250) — matrix operand order repaired and checked by reading, not compiled, not run on hardware
//
#include <hip/hip_runtime.h>

// ---------------------------------------------------------------------------
// Types for CDNA5 WMMA (wave32, v_wmma_f32_16x16x32_bf16)
// ---------------------------------------------------------------------------
typedef __attribute__((ext_vector_type(16))) __bf16 bf16x16;
typedef __attribute__((ext_vector_type(8)))  float  floatx8;
typedef unsigned short u16;

union Frag {
    bf16x16 v;
    uint4   q[2];
};

__device__ __forceinline__ u16 f2bf(float f) {
    unsigned int u = __float_as_uint(f);
    unsigned int r = (u + 0x7FFFu + ((u >> 16) & 1u)) >> 16;
    return (u16)r;
}
__device__ __forceinline__ float bf2f(u16 b) {
    return __uint_as_float(((unsigned int)b) << 16);
}

__device__ __forceinline__ floatx8 wmma_bf16(const Frag& a, const Frag& b, floatx8 c) {
    return __builtin_amdgcn_wmma_f32_16x16x32_bf16(
        /*neg_a=*/false, a.v, /*neg_b=*/false, b.v,
        /*c_mod=*/(short)0, c, /*reuse_a=*/false, /*reuse_b=*/false);
}

// A fragment (16x32 bf16, M x K): lane holds row = lane%16; lanes 0-15 carry
// K=[k0..k0+15]; lanes 16-31 carry K=[k0+16..k0+31].
__device__ __forceinline__ void loadA(const u16* base, int stride, int row, int k0,
                                      int lane, Frag& f) {
    int koff = (lane >> 4) ? 16 : 0;
    const u16* p = base + (size_t)row * stride + k0 + koff;
    f.q[0] = *(const uint4*)(p);
    f.q[1] = *(const uint4*)(p + 8);
}

// B fragment (32x16 bf16, K x N) from row-major weight W[out, in]:
// lane holds column n = lane%16 (= output feature), 16 consecutive K values
// starting at k0 (lanes 0-15) or k0+16 (lanes 16-31).
__device__ __forceinline__ void loadB(const u16* w, int K, int col, int k0,
                                      int lane, Frag& f) {
    int kbase = (lane >> 4) ? 16 : 0;
    const u16* p = w + (size_t)col * K + k0 + kbase;
    f.q[0] = *(const uint4*)(p);
    f.q[1] = *(const uint4*)(p + 8);
}

// ---------------------------------------------------------------------------
// One-time fp32 -> bf16 conversion (relu optionally folded in)
// ---------------------------------------------------------------------------
__global__ void cvt_bf16_kernel(const float* __restrict__ src, u16* __restrict__ dst,
                                int n, int do_relu) {
    int i = blockIdx.x * blockDim.x + threadIdx.x;
    if (i < n) {
        float v = src[i];
        if (do_relu) v = fmaxf(v, 0.0f);
        dst[i] = f2bf(v);
    }
}

// ---------------------------------------------------------------------------
// One GRU time step: h_new = GRUCell(x_t, h_old)
//   x_t rows come from a bf16 embedding table gathered via tokens[b*seqlen+t].
//   Each wave owns one 16x16 (batch x hidden) tile: 4 fp32 accumulators
//   (r, z, gi_n, gh_n), 6 WMMAs per 32-wide K chunk, K=512.
// grid: 128 blocks x 128 threads (4 waves) -> 512 tiles = (B/16)*(D/16)
// ---------------------------------------------------------------------------
__global__ __launch_bounds__(128) void gru_step_kernel(
    const int* __restrict__ tokens, int t, int seqlen,
    const u16* __restrict__ xtab,   // [V, D] bf16 (relu pre-applied for decoder)
    const u16* __restrict__ Wih,    // [3D, D] bf16
    const u16* __restrict__ Whh,    // [3D, D] bf16
    const float* __restrict__ bih, const float* __restrict__ bhh,
    const u16* __restrict__ h_old,  // [B, D] bf16
    u16* __restrict__ h_new)        // [B, D] bf16
{
    const int D = 512;
    int lane = threadIdx.x & 31;
    int wave = threadIdx.x >> 5;
    int tile = blockIdx.x * 4 + wave;   // 0..511
    int n_tile = tile & 31;             // D/16 = 32
    int m_tile = tile >> 5;             // B/16 = 16
    int m_base = m_tile * 16;
    int n_base = n_tile * 16;

    int arow = m_base + (lane & 15);
    int tok  = tokens[arow * seqlen + t];
    int col  = n_base + (lane & 15);

    floatx8 accR = {}, accZ = {}, accGin = {}, accGhn = {};

    for (int k0 = 0; k0 < D; k0 += 32) {
        Frag ax, ah, b;
        loadA(xtab,  D, tok,  k0, lane, ax);
        loadA(h_old, D, arow, k0, lane, ah);

        loadB(Wih, D, col, k0, lane, b);            // r gate, input side
        accR = wmma_bf16(ax, b, accR);
        loadB(Whh, D, col, k0, lane, b);            // r gate, hidden side
        accR = wmma_bf16(ah, b, accR);

        loadB(Wih, D, D + col, k0, lane, b);        // z gate, input side
        accZ = wmma_bf16(ax, b, accZ);
        loadB(Whh, D, D + col, k0, lane, b);        // z gate, hidden side
        accZ = wmma_bf16(ah, b, accZ);

        loadB(Wih, D, 2 * D + col, k0, lane, b);    // n gate, input side
        accGin = wmma_bf16(ax, b, accGin);
        loadB(Whh, D, 2 * D + col, k0, lane, b);    // n gate, hidden side
        accGhn = wmma_bf16(ah, b, accGhn);
    }

    float br   = bih[col]         + bhh[col];
    float bz   = bih[D + col]     + bhh[D + col];
    float bgin = bih[2 * D + col];
    float bghn = bhh[2 * D + col];

    int mrow0 = m_base + ((lane >> 4) << 3);   // C/D layout: vgpr i -> M = i + 8*(lane>=16)
    #pragma unroll
    for (int i = 0; i < 8; ++i) {
        int m = mrow0 + i;
        float r  = 1.0f / (1.0f + __expf(-(accR[i] + br)));
        float z  = 1.0f / (1.0f + __expf(-(accZ[i] + bz)));
        float nn = tanhf(accGin[i] + bgin + r * (accGhn[i] + bghn));
        float ho = bf2f(h_old[(size_t)m * D + col]);
        h_new[(size_t)m * D + col] = f2bf((1.0f - z) * nn + z * ho);
    }
}

// ---------------------------------------------------------------------------
// Decoder output: logits = h @ out_W^T + out_b, then log_softmax over V=512.
// One block per 16 batch rows; 8 waves x 4 col-tiles cover all 512 vocab cols.
// Logits staged in 32 KB LDS, then wave-shuffle max/LSE reduction per row.
// grid: 16 blocks x 256 threads
// ---------------------------------------------------------------------------
__global__ __launch_bounds__(256) void logits_softmax_kernel(
    const u16* __restrict__ h,     // [B, D] bf16
    const u16* __restrict__ Wo,    // [V, D] bf16
    const float* __restrict__ bo,  // [V]
    float* __restrict__ out,       // [B, T, V]
    int t)
{
    const int D = 512, V = 512, T = 256;
    __shared__ float smem[16][512];

    int lane = threadIdx.x & 31;
    int wave = threadIdx.x >> 5;        // 0..7
    int m_base = blockIdx.x * 16;
    int arow = m_base + (lane & 15);

    floatx8 acc[4] = {};
    for (int k0 = 0; k0 < D; k0 += 32) {
        Frag a;
        loadA(h, D, arow, k0, lane, a);
        #pragma unroll
        for (int j = 0; j < 4; ++j) {
            int col = wave * 64 + j * 16 + (lane & 15);
            Frag b;
            loadB(Wo, D, col, k0, lane, b);
            acc[j] = wmma_bf16(a, b, acc[j]);
        }
    }

    int mrow0 = (lane >> 4) << 3;
    #pragma unroll
    for (int j = 0; j < 4; ++j) {
        int col = wave * 64 + j * 16 + (lane & 15);
        float bb = bo[col];
        #pragma unroll
        for (int i = 0; i < 8; ++i)
            smem[mrow0 + i][col] = acc[j][i] + bb;
    }
    __syncthreads();

    // log-softmax: wave w handles rows 2w and 2w+1
    for (int rr = 0; rr < 2; ++rr) {
        int r = wave * 2 + rr;
        float mx = -INFINITY;
        for (int c = lane; c < V; c += 32) mx = fmaxf(mx, smem[r][c]);
        for (int off = 16; off; off >>= 1) mx = fmaxf(mx, __shfl_xor(mx, off, 32));
        float sum = 0.0f;
        for (int c = lane; c < V; c += 32) sum += __expf(smem[r][c] - mx);
        for (int off = 16; off; off >>= 1) sum += __shfl_xor(sum, off, 32);
        float lse = mx + __logf(sum);
        int b = m_base + r;
        float* o = out + ((size_t)b * T + t) * V;
        for (int c = lane; c < V; c += 32) o[c] = smem[r][c] - lse;
    }
}

// ---------------------------------------------------------------------------
// Host launcher
// ---------------------------------------------------------------------------
extern "C" void kernel_launch(void* const* d_in, const int* in_sizes, int n_in,
                              void* d_out, int out_size, void* d_ws, size_t ws_size,
                              hipStream_t stream) {
    const int B = 256, S = 256, T = 256, V = 512, D = 512;

    const int*   src     = (const int*)  d_in[0];
    const int*   trg     = (const int*)  d_in[1];
    // d_in[2], d_in[3]: src_mask / trg_mask (all true) -- unused
    const float* enc_emb = (const float*)d_in[4];
    const float* enc_Wih = (const float*)d_in[5];
    const float* enc_Whh = (const float*)d_in[6];
    const float* enc_bih = (const float*)d_in[7];
    const float* enc_bhh = (const float*)d_in[8];
    const float* dec_emb = (const float*)d_in[9];
    const float* dec_Wih = (const float*)d_in[10];
    const float* dec_Whh = (const float*)d_in[11];
    const float* dec_bih = (const float*)d_in[12];
    const float* dec_bhh = (const float*)d_in[13];
    const float* out_W   = (const float*)d_in[14];
    const float* out_b   = (const float*)d_in[15];
    float* out = (float*)d_out;

    // Carve workspace (256B aligned slices); total ~8.2 MB of bf16 buffers.
    char* wsp = (char*)d_ws;
    auto carve = [&](size_t bytes) -> char* {
        char* p = wsp;
        wsp += (bytes + 255) & ~(size_t)255;
        return p;
    };
    u16* encembb = (u16*)carve((size_t)V * D * 2);
    u16* decembb = (u16*)carve((size_t)V * D * 2);
    u16* eWih    = (u16*)carve((size_t)3 * D * D * 2);
    u16* eWhh    = (u16*)carve((size_t)3 * D * D * 2);
    u16* dWih    = (u16*)carve((size_t)3 * D * D * 2);
    u16* dWhh    = (u16*)carve((size_t)3 * D * D * 2);
    u16* Wo      = (u16*)carve((size_t)V * D * 2);
    u16* hA      = (u16*)carve((size_t)B * D * 2);
    u16* hB      = (u16*)carve((size_t)B * D * 2);

    // One-time bf16 conversions (weights/tables stay hot in 192 MB L2).
    auto cvt = [&](const float* s, u16* d, int n, int relu) {
        cvt_bf16_kernel<<<(n + 255) / 256, 256, 0, stream>>>(s, d, n, relu);
    };
    cvt(enc_emb, encembb, V * D, 0);
    cvt(dec_emb, decembb, V * D, 1);       // relu folded into decoder table
    cvt(enc_Wih, eWih, 3 * D * D, 0);
    cvt(enc_Whh, eWhh, 3 * D * D, 0);
    cvt(dec_Wih, dWih, 3 * D * D, 0);
    cvt(dec_Whh, dWhh, 3 * D * D, 0);
    cvt(out_W,   Wo,   V * D, 0);

    hipMemsetAsync(hA, 0, (size_t)B * D * 2, stream);  // h0 = 0

    u16* hc = hA;
    u16* hn = hB;

    // Encoder: 256 sequential GRU steps (even count -> ends in hA).
    for (int t = 0; t < S; ++t) {
        gru_step_kernel<<<128, 128, 0, stream>>>(src, t, S, encembb, eWih, eWhh,
                                                 enc_bih, enc_bhh, hc, hn);
        u16* tmp = hc; hc = hn; hn = tmp;
    }

    // Decoder: GRU step + fused projection/log-softmax per timestep.
    for (int t = 0; t < T; ++t) {
        gru_step_kernel<<<128, 128, 0, stream>>>(trg, t, T, decembb, dWih, dWhh,
                                                 dec_bih, dec_bhh, hc, hn);
        u16* tmp = hc; hc = hn; hn = tmp;
        logits_softmax_kernel<<<16, 256, 0, stream>>>(hc, Wo, out_b, out, t);
    }
}
